// FlashAttention_43250320671024
// MI455X (gfx1250) — compile-verified, matrix operand order checked by reading, not run on hardware
//
#include <hip/hip_runtime.h>
#include <hip/hip_bf16.h>

// ---------------------------------------------------------------------------
// Causal multi-head attention block, bf16 WMMA pipeline for gfx1250 (MI455X).
//   stage 0: convert x -> bf16 (vectorized), LDS-tiled transpose+convert of
//            w_qkv / w_out -> bf16 [N][K]
//   stage 1: qkv = x @ w_qkv   (32x64 wave tiles, double-buffered fragments,
//            Q pre-scaled, V stored transposed per head)
//   stage 2: flash attention   (online softmax fp32, bf16 WMMA, LDS P-staging,
//            global_prefetch of next key block)
//   stage 3: out = attn @ w_out (32x64 wave tiles, fp32 store)
// ---------------------------------------------------------------------------

#define HIDDEN   2048
#define HEADS    16
#define HEAD_DIM 128
#define SEQ      2048
#define BATCH    2

typedef __bf16 bf16_t;
typedef __attribute__((ext_vector_type(16))) __bf16 v16bf;
typedef __attribute__((ext_vector_type(8)))  __bf16 v8bf;
typedef __attribute__((ext_vector_type(4)))  __bf16 v4bf;
typedef __attribute__((ext_vector_type(8)))  float  v8f;

static __device__ inline bf16_t f2bf(float f) {
    unsigned u = __builtin_bit_cast(unsigned, f);
    unsigned r = u + 0x7FFFu + ((u >> 16) & 1u);      // round-to-nearest-even
    unsigned short h = (unsigned short)(r >> 16);
    return __builtin_bit_cast(bf16_t, h);
}

static __device__ inline v8f vzero8() {
    v8f z;
#pragma unroll
    for (int i = 0; i < 8; ++i) z[i] = 0.0f;
    return z;
}

// Build a 16x32 bf16 A/B fragment's per-lane 16 values from two contiguous
// 8-element (16B) chunks.  Per the CDNA5 16-bit A layout, lane (half = L>>4)
// holds K = half*8 + [0..8) in VGPR0-3 and K = 16 + half*8 + [0..8) in
// VGPR4-7; with a K-contiguous operand both chunks are 16B loads.
static __device__ inline v16bf make_frag(const bf16_t* p0, const bf16_t* p1) {
    v8bf lo = *reinterpret_cast<const v8bf*>(p0);
    v8bf hi = *reinterpret_cast<const v8bf*>(p1);
    v16bf a;
#pragma unroll
    for (int i = 0; i < 8; ++i) { a[i] = lo[i]; a[i + 8] = hi[i]; }
    return a;
}

// -------------------------- stage 0: conversions ---------------------------

__global__ void cvt_f32_bf16(const float* __restrict__ in,
                             bf16_t* __restrict__ out, int n) {
    const float4* in4 = (const float4*)in;
    v4bf* out4 = (v4bf*)out;
    const int n4 = n >> 2;
    for (int i = blockIdx.x * blockDim.x + threadIdx.x; i < n4;
         i += gridDim.x * blockDim.x) {
        float4 v = in4[i];
        v4bf o;
        o[0] = f2bf(v.x); o[1] = f2bf(v.y);
        o[2] = f2bf(v.z); o[3] = f2bf(v.w);
        out4[i] = o;
    }
}

// in: K x N row-major (f32) -> out: N x K row-major (bf16); LDS-tiled 32x32.
// grid (K/32, N/32), 256 threads (32x8 logical).
__global__ void __launch_bounds__(256)
transpose_f32_bf16(const float* __restrict__ in, bf16_t* __restrict__ out,
                   int K, int N) {
    __shared__ float tile[32][33];
    const int tx = threadIdx.x & 31;
    const int ty = threadIdx.x >> 5;          // 0..7
    const int k0 = blockIdx.x * 32;
    const int n0 = blockIdx.y * 32;
#pragma unroll
    for (int r = 0; r < 4; ++r) {
        const int k = k0 + ty + r * 8;
        tile[ty + r * 8][tx] = in[(size_t)k * N + n0 + tx];
    }
    __syncthreads();
#pragma unroll
    for (int r = 0; r < 4; ++r) {
        const int n = n0 + ty + r * 8;
        out[(size_t)n * K + k0 + tx] = f2bf(tile[tx][ty + r * 8]);
    }
}

// ----------------- shared GEMM core: 32x64 tile per wave -------------------
// Double-buffered fragment registers; 8 WMMAs per K-step against 12 16B loads.
template <int KDIM>
static __device__ inline void wave_gemm_32x64(const bf16_t* __restrict__ A,
                                              const bf16_t* __restrict__ Bt,
                                              int m0, int nbase, int lane,
                                              v8f acc[2][4]) {
    const int half = lane >> 4, nl = lane & 15;
    const bf16_t* a0 = A + (size_t)(m0 + nl) * KDIM + half * 8;
    const bf16_t* a1 = a0 + (size_t)16 * KDIM;
    const bf16_t* b0 = Bt + (size_t)(nbase + nl) * KDIM + half * 8;
    const bf16_t* b1 = b0 + (size_t)16 * KDIM;
    const bf16_t* b2 = b0 + (size_t)32 * KDIM;
    const bf16_t* b3 = b0 + (size_t)48 * KDIM;

    v16bf fa[2][2], fb[2][4];
    auto ld = [&](int buf, int kk) {
        fa[buf][0] = make_frag(a0 + kk, a0 + kk + 16);
        fa[buf][1] = make_frag(a1 + kk, a1 + kk + 16);
        fb[buf][0] = make_frag(b0 + kk, b0 + kk + 16);
        fb[buf][1] = make_frag(b1 + kk, b1 + kk + 16);
        fb[buf][2] = make_frag(b2 + kk, b2 + kk + 16);
        fb[buf][3] = make_frag(b3 + kk, b3 + kk + 16);
    };
    ld(0, 0);
#pragma unroll 2
    for (int kk = 0; kk < KDIM; kk += 32) {
        const int cur = (kk >> 5) & 1, nxt = cur ^ 1;
        if (kk + 32 < KDIM) ld(nxt, kk + 32);   // prefetch next K-step
#pragma unroll
        for (int ms = 0; ms < 2; ++ms)
#pragma unroll
            for (int sub = 0; sub < 4; ++sub)
                acc[ms][sub] = __builtin_amdgcn_wmma_f32_16x16x32_bf16(
                    false, fa[cur][ms], false, fb[cur][sub], (short)0,
                    acc[ms][sub], false, false);
    }
}

// ------------------------ stage 1: QKV projection --------------------------
// grid (M/32, N/256), 128 threads (4 waves); each wave owns a 32x64 tile.

__global__ void __launch_bounds__(128)
gemm_qkv(const bf16_t* __restrict__ xb,     // [B*S][HIDDEN]
         const bf16_t* __restrict__ wT,     // [3*HIDDEN][HIDDEN]
         bf16_t* __restrict__ qb,           // [B*H][S][D]  (pre-scaled)
         bf16_t* __restrict__ kbuf,         // [B*H][S][D]
         bf16_t* __restrict__ vtb) {        // [B*H][D][S]  (transposed)
    const int lane = threadIdx.x & 31;
    const int wid  = threadIdx.x >> 5;
    const int half = lane >> 4;
    const int nl   = lane & 15;
    const int m0    = blockIdx.x * 32;
    const int nbase = blockIdx.y * 256 + wid * 64;

    v8f acc[2][4];
#pragma unroll
    for (int ms = 0; ms < 2; ++ms)
#pragma unroll
        for (int s = 0; s < 4; ++s) acc[ms][s] = vzero8();

    wave_gemm_32x64<HIDDEN>(xb, wT, m0, nbase, lane, acc);

    const float qscale = 0.08838834764831845f;  // 1/sqrt(HEAD_DIM)
#pragma unroll
    for (int ms = 0; ms < 2; ++ms) {
#pragma unroll
        for (int sub = 0; sub < 4; ++sub) {
            const int ng = nbase + sub * 16 + nl;
            const int region = ng >> 11;        // 0:Q 1:K 2:V
            const int w = ng & (HIDDEN - 1);
            const int h = w >> 7, d = w & (HEAD_DIM - 1);
#pragma unroll
            for (int r = 0; r < 8; ++r) {
                const int mg = m0 + ms * 16 + r + 8 * half;  // b*S + s
                const int b = mg >> 11, s = mg & (SEQ - 1);
                const size_t bh = (size_t)(b * HEADS + h);
                const float v = acc[ms][sub][r];
                if (region == 0)
                    qb[(bh * SEQ + s) * HEAD_DIM + d] = f2bf(v * qscale);
                else if (region == 1)
                    kbuf[(bh * SEQ + s) * HEAD_DIM + d] = f2bf(v);
                else
                    vtb[(bh * HEAD_DIM + d) * SEQ + s] = f2bf(v);
            }
        }
    }
}

// ------------------------ stage 2: flash attention -------------------------
// grid (S/16/4, B*H), 128 threads (4 waves); one wave per 16-query tile.

__global__ void __launch_bounds__(128)
attn_kernel(const bf16_t* __restrict__ qb,
            const bf16_t* __restrict__ kbuf,
            const bf16_t* __restrict__ vtb,
            bf16_t* __restrict__ attn) {     // [B*S][HIDDEN]
    __shared__ bf16_t pstage[4][16 * 32];
    const int lane = threadIdx.x & 31;
    const int wid  = threadIdx.x >> 5;
    const int half = lane >> 4;
    const int nl   = lane & 15;
    const int qtile = blockIdx.x * 4 + wid;   // 0..127
    const int bh    = blockIdx.y;             // 0..31

    const bf16_t* Q = qb   + (size_t)bh * SEQ * HEAD_DIM;
    const bf16_t* K = kbuf + (size_t)bh * SEQ * HEAD_DIM;
    const bf16_t* V = vtb  + (size_t)bh * HEAD_DIM * SEQ;
    bf16_t* pl = pstage[wid];

    // Q fragments for all 4 d-chunks (16 rows x 128) stay in registers.
    v16bf qf[4];
    {
        const bf16_t* qrow = Q + (size_t)(qtile * 16 + nl) * HEAD_DIM;
#pragma unroll
        for (int dc = 0; dc < 4; ++dc) {
            const bf16_t* p = qrow + dc * 32 + half * 8;
            qf[dc] = make_frag(p, p + 16);
        }
    }

    v8f o[8];
    float mrow[8], lrow[8];
#pragma unroll
    for (int c = 0; c < 8; ++c) o[c] = vzero8();
#pragma unroll
    for (int r = 0; r < 8; ++r) { mrow[r] = -3.0e38f; lrow[r] = 0.0f; }

    const int qbase = qtile * 16;
    const int nb = (qbase + 16 + 31) >> 5;     // 32-key blocks (causal limit)

    for (int kb = 0; kb < nb; ++kb) {
        // ---- prefetch next key block (K rows + Vt columns) into cache ----
        if (kb + 1 < nb) {
            const bf16_t* kn = K + (size_t)((kb + 1) * 32 + lane) * HEAD_DIM;
            __builtin_prefetch(kn, 0, 3);
            __builtin_prefetch(kn + 64, 0, 3);
#pragma unroll
            for (int i = 0; i < 4; ++i)
                __builtin_prefetch(
                    V + (size_t)(lane + 32 * i) * SEQ + (kb + 1) * 32, 0, 3);
        }

        // ---- scores: two 16x16 tiles over 32 keys, K-dim = 128 -----------
        v8f s0 = vzero8(), s1 = vzero8();
#pragma unroll
        for (int dc = 0; dc < 4; ++dc) {
            const bf16_t* k0 =
                K + (size_t)(kb * 32 + nl) * HEAD_DIM + dc * 32 + half * 8;
            v16bf b0 = make_frag(k0, k0 + 16);
            s0 = __builtin_amdgcn_wmma_f32_16x16x32_bf16(
                false, qf[dc], false, b0, (short)0, s0, false, false);
            const bf16_t* k1 =
                K + (size_t)(kb * 32 + 16 + nl) * HEAD_DIM + dc * 32 + half * 8;
            v16bf b1 = make_frag(k1, k1 + 16);
            s1 = __builtin_amdgcn_wmma_f32_16x16x32_bf16(
                false, qf[dc], false, b1, (short)0, s1, false, false);
        }

        // ---- causal mask + online softmax (fp32, per-row across halves) --
        const int key0 = kb * 32 + nl, key1 = key0 + 16;
#pragma unroll
        for (int r = 0; r < 8; ++r) {
            const int qr = qbase + r + 8 * half;
            float a0 = (key0 <= qr) ? s0[r] : -1.0e30f;
            float a1 = (key1 <= qr) ? s1[r] : -1.0e30f;
            float vmax = fmaxf(a0, a1);
#pragma unroll
            for (int off = 8; off >= 1; off >>= 1)
                vmax = fmaxf(vmax, __shfl_xor(vmax, off, 16));
            const float mnew  = fmaxf(mrow[r], vmax);
            const float alpha = __expf(mrow[r] - mnew);
            mrow[r] = mnew;
            const float p0 = __expf(a0 - mnew);
            const float p1 = __expf(a1 - mnew);
            float rs = p0 + p1;
#pragma unroll
            for (int off = 8; off >= 1; off >>= 1)
                rs += __shfl_xor(rs, off, 16);
            lrow[r] = alpha * lrow[r] + rs;
#pragma unroll
            for (int c = 0; c < 8; ++c) o[c][r] *= alpha;
            // stage P (C-layout) into LDS as bf16 for the A-layout reload
            pl[(r + 8 * half) * 32 + nl]      = f2bf(p0);
            pl[(r + 8 * half) * 32 + 16 + nl] = f2bf(p1);
        }
        // DS ops from one wave are in-order; the wait + memory clobber stop
        // the compiler from hoisting the reload above the stores.
        asm volatile("s_wait_dscnt 0" ::: "memory");

        const bf16_t* pp = pl + nl * 32 + half * 8;
        v16bf pf = make_frag(pp, pp + 16);

        // ---- O += P @ V  (Vt[d][s] -> contiguous B-fragments) ------------
#pragma unroll
        for (int c = 0; c < 8; ++c) {
            const bf16_t* vrow =
                V + (size_t)(c * 16 + nl) * SEQ + kb * 32 + half * 8;
            v16bf vf = make_frag(vrow, vrow + 16);
            o[c] = __builtin_amdgcn_wmma_f32_16x16x32_bf16(
                false, pf, false, vf, (short)0, o[c], false, false);
        }
        asm volatile("s_wait_dscnt 0" ::: "memory");  // reads done before next overwrite
    }

    // ---- epilogue: O / l, store to [b][s][h*128+d] ------------------------
    const int b = bh >> 4, h = bh & (HEADS - 1);
#pragma unroll
    for (int r = 0; r < 8; ++r) {
        const float inv = 1.0f / lrow[r];
        const int sq = qbase + r + 8 * half;
        const size_t row = ((size_t)(b * SEQ + sq)) * HIDDEN + h * HEAD_DIM;
#pragma unroll
        for (int c = 0; c < 8; ++c)
            attn[row + c * 16 + nl] = f2bf(o[c][r] * inv);
    }
}

// ------------------------- stage 3: output GEMM ----------------------------
// grid (M/32, N/256), 128 threads (4 waves); each wave owns a 32x64 tile.

__global__ void __launch_bounds__(128)
gemm_out(const bf16_t* __restrict__ a,      // [B*S][HIDDEN]
         const bf16_t* __restrict__ wT,     // [HIDDEN][HIDDEN] (transposed)
         float* __restrict__ out) {         // [B*S][HIDDEN]
    const int lane = threadIdx.x & 31;
    const int wid  = threadIdx.x >> 5;
    const int half = lane >> 4;
    const int nl   = lane & 15;
    const int m0    = blockIdx.x * 32;
    const int nbase = blockIdx.y * 256 + wid * 64;

    v8f acc[2][4];
#pragma unroll
    for (int ms = 0; ms < 2; ++ms)
#pragma unroll
        for (int s = 0; s < 4; ++s) acc[ms][s] = vzero8();

    wave_gemm_32x64<HIDDEN>(a, wT, m0, nbase, lane, acc);

#pragma unroll
    for (int ms = 0; ms < 2; ++ms)
#pragma unroll
        for (int sub = 0; sub < 4; ++sub) {
            const int ng = nbase + sub * 16 + nl;
#pragma unroll
            for (int r = 0; r < 8; ++r) {
                const int mg = m0 + ms * 16 + r + 8 * half;
                out[(size_t)mg * HIDDEN + ng] = acc[ms][sub][r];
            }
        }
}

// ---------------------------------------------------------------------------

extern "C" void kernel_launch(void* const* d_in, const int* in_sizes, int n_in,
                              void* d_out, int out_size, void* d_ws,
                              size_t ws_size, hipStream_t stream) {
    const float* x     = (const float*)d_in[0];   // [2,2048,2048]
    const float* w_qkv = (const float*)d_in[1];   // [2048,6144]
    const float* w_out = (const float*)d_in[2];   // [2048,2048]
    float* out = (float*)d_out;

    char* ws = (char*)d_ws;
    size_t off = 0;
    auto carve = [&](size_t bytes) {
        void* p = ws + off;
        off += (bytes + 255) & ~(size_t)255;
        return p;
    };
    const size_t nx  = (size_t)BATCH * SEQ * HIDDEN;
    bf16_t* xb    = (bf16_t*)carve(nx * 2);                          // 16.8 MB
    bf16_t* wqkvT = (bf16_t*)carve((size_t)3 * HIDDEN * HIDDEN * 2); // 25.2 MB
    bf16_t* woutT = (bf16_t*)carve((size_t)HIDDEN * HIDDEN * 2);     //  8.4 MB
    bf16_t* qb    = (bf16_t*)carve(nx * 2);                          // 16.8 MB
    bf16_t* kbuf  = (bf16_t*)carve(nx * 2);                          // 16.8 MB
    bf16_t* vtb   = (bf16_t*)carve(nx * 2);                          // 16.8 MB
    bf16_t* attnb = (bf16_t*)carve(nx * 2);                          // 16.8 MB
    (void)ws_size;

    // stage 0: precision conversion + weight transposes
    cvt_f32_bf16<<<1024, 256, 0, stream>>>(x, xb, (int)nx);
    transpose_f32_bf16<<<dim3(HIDDEN / 32, 3 * HIDDEN / 32), 256, 0, stream>>>(
        w_qkv, wqkvT, HIDDEN, 3 * HIDDEN);
    transpose_f32_bf16<<<dim3(HIDDEN / 32, HIDDEN / 32), 256, 0, stream>>>(
        w_out, woutT, HIDDEN, HIDDEN);

    // stage 1: qkv projection  (M=4096, N=6144, K=2048)
    gemm_qkv<<<dim3(128, 24), 128, 0, stream>>>(xb, wqkvT, qb, kbuf, vtb);

    // stage 2: causal flash attention
    attn_kernel<<<dim3(32, 32), 128, 0, stream>>>(qb, kbuf, vtb, attnb);

    // stage 3: output projection (M=4096, N=2048, K=2048), fp32 out
    gemm_out<<<dim3(128, 8), 128, 0, stream>>>(attnb, woutT, out);
}
